// MarketRegimeHMM_8435315769985
// MI455X (gfx1250) — compile-verified, matrix operand order checked by reading, not run on hardware
//
#include <hip/hip_runtime.h>
#include <hip/hip_bf16.h>
#include <math.h>

// ---------------------------------------------------------------------------
// IO-HMM forward pass for MI455X (gfx1250, wave32).
//
// Kernel 1 (hmm_stage1): parallel over all B*T rows in 16-row tiles, one tile
//   per wave. Uses v_wmma_f32_16x16x32_f16 for
//     (a) Gaussian emission log-probs   [16x32]·[32x16] (K = [obs^2 | obs])
//     (b) transition-net layer 1        K=4 padded, 4 col tiles (50 hidden)
//     (c) transition-net layer 2        block-diag as dense zero-masked GEMM,
//                                       K=50 padded to 64 (2 K-steps x 2 tiles)
//   tanh uses the native gfx1250 v_tanh_f32 (branch-free) when available.
//   Then softmax+log per source state and stores the fused scan operator
//     M'[b,t-1,i,j] = log_softmax(logits)[i,j] + log_emis[b,t,j]
//   to workspace; writes alpha0 for t==0 directly to the output.
//
// Kernel 2 (hmm_scan): one wave per batch chain. Lane l<25 holds (i,j)=
//   (l/5,l%5); the 5-way logsumexp contraction is done with wave32 shuffles,
//   with the next step's M' load software-pipelined and a deep
//   global_prefetch_b8 stream to keep the serial consumer fed from L2.
// ---------------------------------------------------------------------------

typedef __attribute__((ext_vector_type(16))) _Float16 v16h;
typedef __attribute__((ext_vector_type(8)))  _Float16 v8h;
typedef __attribute__((ext_vector_type(8)))  float    v8f;

#define NS 5
#define ND 16
#define NF 4
#define NH 10
#define NB 256
#define NT 4096
#define LOG_2PI 1.8378770664093453f

#define WAVES_PER_BLOCK 8
#define S1_BLOCKS 1024   // 8192 waves, 65536 tiles -> exactly 8 tiles per wave

__device__ __forceinline__ int kmap(int e, int hi) {
    // 16-bit A/B operand lane packing (CDNA5 ISA):
    // lanes 0-15: VGPR0-3 -> K=0..7, VGPR4-7 -> K=16..23; lanes 16-31: +8.
    return (e < 8 ? e : e + 8) + (hi ? 8 : 0);
}

__device__ __forceinline__ v8f wmma16(v16h a, v16h b, v8f c) {
    return __builtin_amdgcn_wmma_f32_16x16x32_f16(
        /*neg_a=*/false, a, /*neg_b=*/false, b,
        /*c_mod=*/(short)0, c, /*reuse_a=*/false, /*reuse_b=*/false);
}

__device__ __forceinline__ v8f splat8(float x) {
    v8f c;
#pragma unroll
    for (int r = 0; r < 8; ++r) c[r] = x;
    return c;
}

// Branch-free tanh: native gfx1250 v_tanh_f32 if the builtin exists,
// otherwise tanh(x) = 1 - 2/(e^{2x}+1) via v_exp_f32 (no exec-mask splits).
__device__ __forceinline__ float fast_tanh(float x) {
#if __has_builtin(__builtin_amdgcn_tanhf)
    return __builtin_amdgcn_tanhf(x);
#else
    float e = __expf(2.0f * x);
    return 1.0f - 2.0f / (e + 1.0f);
#endif
}

__global__ __launch_bounds__(256)
void hmm_stage1(const float* __restrict__ obs,  const float* __restrict__ feat,
                const float* __restrict__ means, const float* __restrict__ logv,
                const float* __restrict__ log_init, const float* __restrict__ tbias,
                const float* __restrict__ W1, const float* __restrict__ b1,
                const float* __restrict__ W2, const float* __restrict__ b2,
                float* __restrict__ out_alpha, float* __restrict__ Mws)
{
    __shared__ _Float16 sh_h[WAVES_PER_BLOCK][16 * 64];   // tanh(hidden), f16
    __shared__ float    sh_le[WAVES_PER_BLOCK][16 * 8];   // emission logp
    __shared__ float    sh_lg[WAVES_PER_BLOCK][16 * 32];  // logits

    const int tid  = threadIdx.x;
    const int lane = tid & 31;
    const int w    = tid >> 5;
    const int hi   = lane >> 4;
    const int n    = lane & 15;   // column (B/C/D) or row (A) index of this lane

    // ---------------- loop-invariant B fragments (held in VGPRs) -----------
    // Emission:  le = A * B + c,  A[m][k<16]=obs^2, A[m][k>=16]=obs,
    //   B[k<16][s] = -0.5*inv_v[s][k],  B[k>=16][s] = means*inv_v,  c = const.
    v16h Bem;
    float cem = 0.f;
    if (n < NS) {
        float s = 0.f;
        for (int d = 0; d < ND; ++d) {
            float lv = logv[n * ND + d];
            float mu = means[n * ND + d];
            s += lv + mu * mu * __expf(-lv);
        }
        cem = -0.5f * (ND * LOG_2PI + s);
    }
#pragma unroll
    for (int e = 0; e < 16; ++e) {
        int k = kmap(e, hi);
        float val = 0.f;
        if (n < NS) {
            if (k < 16) {
                val = -0.5f * __expf(-logv[n * ND + k]);
            } else {
                int d = k - 16;
                val = means[n * ND + d] * __expf(-logv[n * ND + d]);
            }
        }
        Bem[e] = (_Float16)val;
    }

    // Layer 1: columns c = i*10+h (50 valid of 64), K = 4 features (zero pad).
    v16h Bw1[4];
    float cw1[4];
#pragma unroll
    for (int t4 = 0; t4 < 4; ++t4) {
        int c = 16 * t4 + n;
        bool cv = (c < NS * NH);
        int i = c / NH, h = c % NH;
        cw1[t4] = cv ? b1[i * NH + h] : 0.f;
#pragma unroll
        for (int e = 0; e < 16; ++e) {
            int k = kmap(e, hi);
            float val = (cv && k < NF) ? W1[(i * NH + h) * NF + k] : 0.f;
            Bw1[t4][e] = (_Float16)val;
        }
    }

    // Layer 2: dense zero-masked block-diagonal; rows c = i'*10+h (50 of 64),
    // columns j = i*5+o (25 of 32). Nonzero only when i'==i.
    v16h Bw2[2][2];
    float clg[2];
#pragma unroll
    for (int ct = 0; ct < 2; ++ct) {
        int j = 16 * ct + n;
        bool jv = (j < NS * NS);
        int i = j / NS, o = j % NS;
        clg[ct] = jv ? (b2[i * NS + o] + tbias[i * NS + o]) : 0.f;
#pragma unroll
        for (int kt = 0; kt < 2; ++kt) {
#pragma unroll
            for (int e = 0; e < 16; ++e) {
                int c = kt * 32 + kmap(e, hi);
                float val = 0.f;
                if (jv && c < NS * NH) {
                    int ii = c / NH, hh = c % NH;
                    if (ii == i) val = W2[i * NS * NH + o * NH + hh];
                }
                Bw2[ct][kt][e] = (_Float16)val;
            }
        }
    }

    const int nTiles = (NB * NT) / 16;
    const int waveGlobal = blockIdx.x * WAVES_PER_BLOCK + w;
    const int nWaves = gridDim.x * WAVES_PER_BLOCK;

    for (int tile = waveGlobal; tile < nTiles; tile += nWaves) {
        const int rowbase = tile * 16;            // tiles never span batches
        const int bIdx = rowbase / NT;
        const int tbase = rowbase % NT;

        // ---- emission ----
        const float* op = obs + (size_t)(rowbase + n) * ND + hi * 8;
        float ov[8];
#pragma unroll
        for (int e = 0; e < 8; ++e) ov[e] = op[e];
        v16h Aem;
#pragma unroll
        for (int e = 0; e < 8; ++e) {
            Aem[e]     = (_Float16)(ov[e] * ov[e]);
            Aem[e + 8] = (_Float16)ov[e];
        }
        v8f le = wmma16(Aem, Bem, splat8(cem));

        // ---- transition net, layer 1 ----
        v16h Atr;
#pragma unroll
        for (int e = 0; e < 16; ++e) Atr[e] = (_Float16)0.f;
        if (!hi) {  // lanes 0-15 carry K=0..7; features live at K<4
            const float* fp = feat + (size_t)(rowbase + n) * NF;
#pragma unroll
            for (int e = 0; e < NF; ++e) Atr[e] = (_Float16)fp[e];
        }
        v8f hp[4];
#pragma unroll
        for (int t4 = 0; t4 < 4; ++t4)
            hp[t4] = wmma16(Atr, Bw1[t4], splat8(cw1[t4]));

        // tanh (native v_tanh_f32) + stage results in this wave's LDS slice
#pragma unroll
        for (int t4 = 0; t4 < 4; ++t4)
#pragma unroll
            for (int r = 0; r < 8; ++r) {
                int m = r + hi * 8;
                sh_h[w][m * 64 + 16 * t4 + n] = (_Float16)fast_tanh(hp[t4][r]);
            }
        if (n < NS)
#pragma unroll
            for (int r = 0; r < 8; ++r)
                sh_le[w][(r + hi * 8) * 8 + n] = le[r];
        __syncthreads();

        // ---- layer 2: rebuild A fragments from LDS (two b128 loads each) ----
        v16h A2[2];
#pragma unroll
        for (int kt = 0; kt < 2; ++kt) {
            const _Float16* hpB = &sh_h[w][n * 64 + kt * 32 + hi * 8];
            v8h lo = *(const v8h*)hpB;          // K-halves e=0..7
            v8h hi8 = *(const v8h*)(hpB + 16);  // K-halves e=8..15
#pragma unroll
            for (int e = 0; e < 8; ++e) { A2[kt][e] = lo[e]; A2[kt][e + 8] = hi8[e]; }
        }
        v8f lg[2];
#pragma unroll
        for (int ct = 0; ct < 2; ++ct) {
            v8f acc = wmma16(A2[0], Bw2[ct][0], splat8(clg[ct]));
            lg[ct] = wmma16(A2[1], Bw2[ct][1], acc);
        }
#pragma unroll
        for (int ct = 0; ct < 2; ++ct)
#pragma unroll
            for (int r = 0; r < 8; ++r)
                sh_lg[w][(r + hi * 8) * 32 + ct * 16 + n] = lg[ct][r];
        __syncthreads();

        // ---- softmax rows, fuse emission, emit scan operator M' ----
        for (int u = lane; u < 16 * NS; u += 32) {
            int m = u / NS, i = u % NS;
            int t = tbase + m;
            float l[5];
#pragma unroll
            for (int o = 0; o < 5; ++o) l[o] = sh_lg[w][m * 32 + i * 5 + o];
            float mx = fmaxf(fmaxf(fmaxf(l[0], l[1]), fmaxf(l[2], l[3])), l[4]);
            float ssum = 0.f;
#pragma unroll
            for (int o = 0; o < 5; ++o) ssum += __expf(l[o] - mx);
            float lse = mx + __logf(ssum);
            float lev[5];
#pragma unroll
            for (int o = 0; o < 5; ++o) lev[o] = sh_le[w][m * 8 + o];

            if (t == 0) {
                if (i == 0) {  // alpha0 = log_softmax(log_initial) + le0
                    float li[5];
#pragma unroll
                    for (int q = 0; q < 5; ++q) li[q] = log_init[q];
                    float m2 = fmaxf(fmaxf(fmaxf(li[0], li[1]), fmaxf(li[2], li[3])), li[4]);
                    float s2 = 0.f;
#pragma unroll
                    for (int q = 0; q < 5; ++q) s2 += __expf(li[q] - m2);
                    float lz = m2 + __logf(s2);
#pragma unroll
                    for (int q = 0; q < 5; ++q)
                        out_alpha[(size_t)bIdx * NT * NS + q] = (li[q] - lz) + lev[q];
                }
            } else {
                size_t base = ((size_t)bIdx * (NT - 1) + (t - 1)) * 25 + i * 5;
#pragma unroll
                for (int o = 0; o < 5; ++o) {
                    float p = __expf(l[o] - lse);
                    Mws[base + o] = __logf(p + 1e-10f) + lev[o];
                }
            }
        }
        __syncthreads();
    }
}

// ---------------------------------------------------------------------------
// Sequential forward scan: one wave per batch chain, S=5 states.
// ---------------------------------------------------------------------------
#define SCAN_PF 24   // prefetch distance in scan steps (24*100B = 2.4KB ahead)

__global__ __launch_bounds__(128)
void hmm_scan(const float* __restrict__ Mws, float* __restrict__ ll,
              float* __restrict__ out_alpha)
{
    const int lane = threadIdx.x & 31;
    const int w = threadIdx.x >> 5;
    const int b = blockIdx.x * 4 + w;
    const int j = lane % 5;           // destination state
    const int i = lane / 5;           // source state (lanes < 25)
    const bool act = lane < 25;

    // every lane mirrors alpha[lane%5]; shfl(alpha, i) then yields alpha[i]
    float alpha = out_alpha[(size_t)b * NT * NS + j];
    const float* Mb = Mws + (size_t)b * (NT - 1) * 25;

    float mcur = act ? Mb[lane] : -1e30f;   // operator for step t=1
    for (int t = 1; t < NT; ++t) {
        // software-pipeline the next step's operator load
        float mnext = (t < NT - 1 && act) ? Mb[(size_t)t * 25 + lane] : -1e30f;
        // deep prefetch of the operator stream (global_prefetch_b8)
        if (t + SCAN_PF < NT - 1)
            __builtin_prefetch(Mb + (size_t)(t + SCAN_PF) * 25 + lane, 0, 0);

        float ai = __shfl(alpha, i, 32);
        float v = act ? (ai + mcur) : -1e30f;
        float v0 = __shfl(v, j,      32);
        float v1 = __shfl(v, j + 5,  32);
        float v2 = __shfl(v, j + 10, 32);
        float v3 = __shfl(v, j + 15, 32);
        float v4 = __shfl(v, j + 20, 32);
        float mx = fmaxf(fmaxf(fmaxf(v0, v1), fmaxf(v2, v3)), v4);
        float s = __expf(v0 - mx) + __expf(v1 - mx) + __expf(v2 - mx) +
                  __expf(v3 - mx) + __expf(v4 - mx);
        alpha = mx + __logf(s);   // le already fused into M'

        if (lane < 5) out_alpha[((size_t)b * NT + t) * NS + lane] = alpha;
        mcur = mnext;
    }

    float a0 = __shfl(alpha, 0, 32), a1 = __shfl(alpha, 1, 32),
          a2 = __shfl(alpha, 2, 32), a3 = __shfl(alpha, 3, 32),
          a4 = __shfl(alpha, 4, 32);
    float mx = fmaxf(fmaxf(fmaxf(a0, a1), fmaxf(a2, a3)), a4);
    float s = __expf(a0 - mx) + __expf(a1 - mx) + __expf(a2 - mx) +
              __expf(a3 - mx) + __expf(a4 - mx);
    if (lane == 0) ll[b] = mx + __logf(s);
}

extern "C" void kernel_launch(void* const* d_in, const int* in_sizes, int n_in,
                              void* d_out, int out_size, void* d_ws, size_t ws_size,
                              hipStream_t stream) {
    (void)in_sizes; (void)n_in; (void)out_size; (void)ws_size;
    const float* obs      = (const float*)d_in[0];
    const float* feat     = (const float*)d_in[1];
    const float* means    = (const float*)d_in[2];
    const float* logv     = (const float*)d_in[3];
    const float* log_init = (const float*)d_in[4];
    const float* tbias    = (const float*)d_in[5];
    const float* W1       = (const float*)d_in[6];
    const float* b1       = (const float*)d_in[7];
    const float* W2       = (const float*)d_in[8];
    const float* b2       = (const float*)d_in[9];

    float* out   = (float*)d_out;
    float* ll    = out;             // [B]
    float* alpha = out + NB;        // [B, T, S]
    float* Mws   = (float*)d_ws;    // [B, T-1, 5, 5] fused scan operators (~100 MB)

    hmm_stage1<<<S1_BLOCKS, 256, 0, stream>>>(obs, feat, means, logv, log_init,
                                              tbias, W1, b1, W2, b2, alpha, Mws);
    hmm_scan<<<NB / 4, 128, 0, stream>>>(Mws, ll, alpha);
}
